// DiffuMamba3_43636867728068
// MI455X (gfx1250) — matrix operand order repaired and checked by reading, not compiled, not run on hardware
//
#include <hip/hip_runtime.h>
#include <math.h>

// ---------------- model dims ----------------
#define B_    2
#define L_    512
#define D_    512
#define COND_ 128
#define DI_   1024
#define DS_   16
#define DT_   32
#define NH_   8
#define HD_   64
#define HID_  1024
#define V_    50304
#define FREQ_ 256

typedef __attribute__((ext_vector_type(16))) _Float16 v16h;
typedef __attribute__((ext_vector_type(8)))  _Float16 v8h;
typedef __attribute__((ext_vector_type(4)))  _Float16 v4h;
typedef __attribute__((ext_vector_type(8)))  float    v8f;
typedef __attribute__((ext_vector_type(4)))  float    v4f;

// =====================================================================
// Tiled WMMA GEMM:  C[M,N] = alpha * A[M,K] @ B(K,N)
//  - A row-major (stride lda); B is [K,N] (TRANSB=0) or [N,K] (TRANSB=1)
//  - batched via grid.z (outer/inner split for per-(b,h) attention)
//  - fp32 in/out; fp16 WMMA math, fp32 accumulate
// Block = 256 threads = 8 waves. Block tile 128(M) x 64(N), K step 32.
// Wave w owns M rows [w*16, w*16+16) and ALL four 16-wide N tiles:
// 4 accumulators -> 4 v_wmma per K step sharing one A fragment.
// Double-buffered LDS; staging split into load-all / store-all phases so
// global loads issue as one clause; all fragments loaded before the WMMA
// chain so the 4 matrix ops issue back-to-back.
// Requires: M % 128 == 0, N % 64 == 0, K % 32 == 0 (true for all uses).
// =====================================================================
template<int TRANSB>
__global__ __launch_bounds__(256)
void wmma_gemm_kernel(const float* __restrict__ A, const float* __restrict__ Bg,
                      float* __restrict__ C,
                      int K, int lda, int ldb, int ldc, float alpha,
                      int inner,
                      long offA, long offB, long offC,
                      long sAo, long sAi, long sBo, long sBi, long sCo, long sCi)
{
  __shared__ __align__(16) _Float16 As[2][128][32];  // [buf][m][k]
  __shared__ __align__(16) _Float16 Bs[2][64][32];   // [buf][n][k] (holds B[k][n])

  const int tid = threadIdx.x;
  const int z  = blockIdx.z;
  const int bo = z / inner, bi = z - bo * inner;
  const float* Ap = A + offA + (long)bo * sAo + (long)bi * sAi;
  const float* Bp = Bg + offB + (long)bo * sBo + (long)bi * sBi;
  float*       Cp = C + offC + (long)bo * sCo + (long)bi * sCi;

  const int row0 = blockIdx.y * 128;
  const int col0 = blockIdx.x * 64;

  const int w    = tid >> 5;        // wave id: M row-block
  const int lane = tid & 31;
  const int lr   = lane & 15;
  const int khi  = lane >> 4;

  // per-thread chunk coordinates (loop-invariant)
  int am[4], ak[4];                 // A: 4 chunks of float4
  #pragma unroll
  for (int j = 0; j < 4; ++j) {
    int cidx = tid + j * 256;
    am[j] = cidx >> 3; ak[j] = (cidx & 7) * 4;
  }
  int bn[2], bk[2];                 // B: 2 chunks
  #pragma unroll
  for (int j = 0; j < 2; ++j) {
    int cidx = tid + j * 256;
    if (TRANSB) { bn[j] = cidx >> 3; bk[j] = (cidx & 7) * 4; }
    else        { bn[j] = cidx & 63; bk[j] = (cidx >> 6) * 4; }
  }

  // stage one 128x32 A tile + 64x32 B tile (f32 -> f16) into buffer `buf`.
  // Phase 1: issue ALL global loads; Phase 2: convert + LDS stores.
  auto stage = [&](int buf, int k0) {
    v4f fa[4], fb[2];
    #pragma unroll
    for (int j = 0; j < 4; ++j)
      fa[j] = *(const v4f*)&Ap[(long)(row0 + am[j]) * lda + (k0 + ak[j])];
    if (TRANSB) {
      #pragma unroll
      for (int j = 0; j < 2; ++j)
        fb[j] = *(const v4f*)&Bp[(long)(col0 + bn[j]) * ldb + (k0 + bk[j])];
    } else {
      #pragma unroll
      for (int j = 0; j < 2; ++j)
        #pragma unroll
        for (int q = 0; q < 4; ++q)
          fb[j][q] = Bp[(long)(k0 + bk[j] + q) * ldb + (col0 + bn[j])];
    }
    #pragma unroll
    for (int j = 0; j < 4; ++j) {
      v4h hv;
      #pragma unroll
      for (int q = 0; q < 4; ++q) hv[q] = (_Float16)fa[j][q];
      *(v4h*)&As[buf][am[j]][ak[j]] = hv;            // ds_store_b64
    }
    #pragma unroll
    for (int j = 0; j < 2; ++j) {
      v4h hv;
      #pragma unroll
      for (int q = 0; q < 4; ++q) hv[q] = (_Float16)fb[j][q];
      *(v4h*)&Bs[buf][bn[j]][bk[j]] = hv;
    }
  };

  v8f acc[4] = {v8f{}, v8f{}, v8f{}, v8f{}};

  stage(0, 0);
  for (int k0 = 0; k0 < K; k0 += 32) {
    const int buf = (k0 >> 5) & 1;
    __syncthreads();                 // buf staged; prev reads of buf^1 done
    if (k0 + 32 < K) stage(buf ^ 1, k0 + 32);   // prefetch next tile

    // Load A fragment + ALL 4 B fragments first (one dscnt wait),
    // then issue the 4 WMMAs back-to-back.
    v8h alo = *(const v8h*)&As[buf][w * 16 + lr][khi * 8];
    v8h ahi = *(const v8h*)&As[buf][w * 16 + lr][16 + khi * 8];
    v16h av;
    #pragma unroll
    for (int i = 0; i < 8; ++i) { av[i] = alo[i]; av[i + 8] = ahi[i]; }

    v16h bv[4];
    #pragma unroll
    for (int nb = 0; nb < 4; ++nb) {
      v8h blo = *(const v8h*)&Bs[buf][nb * 16 + lr][khi * 8];
      v8h bhi = *(const v8h*)&Bs[buf][nb * 16 + lr][16 + khi * 8];
      #pragma unroll
      for (int i = 0; i < 8; ++i) { bv[nb][i] = blo[i]; bv[nb][i + 8] = bhi[i]; }
    }

    #pragma unroll
    for (int nb = 0; nb < 4; ++nb)
      acc[nb] = __builtin_amdgcn_wmma_f32_16x16x32_f16(false, av, false, bv[nb],
                                                       (short)0, acc[nb], false, false);
  }

  // epilogue: VGPR r holds M = 8*khi + r, N = lr
  #pragma unroll
  for (int nb = 0; nb < 4; ++nb) {
    #pragma unroll
    for (int r = 0; r < 8; ++r) {
      int m = row0 + w * 16 + khi * 8 + r;
      int n = col0 + nb * 16 + lr;
      Cp[(long)m * ldc + n] = acc[nb][r] * alpha;
    }
  }
}

// =====================================================================
// Elementwise / small kernels
// =====================================================================
__global__ void embed_kernel(const int* __restrict__ ids,
                             const float* __restrict__ tok,
                             const float* __restrict__ pos,
                             float* __restrict__ x)
{
  int bt = blockIdx.x;               // 0..B*L-1
  int t  = bt % L_;
  int token = ids[bt];
  for (int d = threadIdx.x; d < D_; d += blockDim.x)
    x[(long)bt * D_ + d] = tok[(long)token * D_ + d] + pos[(long)t * D_ + d];
}

__global__ void timestep_kernel(const float* __restrict__ t,
                                const float* __restrict__ w1, const float* __restrict__ b1,
                                const float* __restrict__ w2, const float* __restrict__ b2,
                                float* __restrict__ c)
{
  __shared__ float emb[FREQ_];
  __shared__ float h1[COND_];
  int b = blockIdx.x;
  int j = threadIdx.x;               // 0..127
  float tv = t[b];
  for (int i = j; i < FREQ_; i += COND_) {
    int half = FREQ_ / 2;
    int fi = (i < half) ? i : (i - half);
    float freq = __expf(-logf(10000.f) * (float)fi / (float)half);
    float arg = tv * freq;
    emb[i] = (i < half) ? cosf(arg) : sinf(arg);
  }
  __syncthreads();
  float acc = b1[j];
  for (int i = 0; i < FREQ_; ++i) acc += emb[i] * w1[i * COND_ + j];
  h1[j] = acc / (1.f + __expf(-acc));      // silu
  __syncthreads();
  float acc2 = b2[j];
  for (int i = 0; i < COND_; ++i) acc2 += h1[i] * w2[i * COND_ + j];
  c[b * COND_ + j] = acc2;
}

// mod[b, :] = c[b] @ W[COND,3D] + bias   (M=2 -> scalar kernel)
__global__ void mod_kernel(const float* __restrict__ c, const float* __restrict__ w,
                           const float* __restrict__ bias, float* __restrict__ mod)
{
  int b = blockIdx.y;
  int n = blockIdx.x * 256 + threadIdx.x;
  float acc = bias[n];
  for (int k = 0; k < COND_; ++k) acc += c[b * COND_ + k] * w[k * (3 * D_) + n];
  mod[b * (3 * D_) + n] = acc;
}

// h = ln(x) * (1 + scale) + shift     (mod = [shift | scale | gate])
__global__ void adaln_kernel(const float* __restrict__ x, const float* __restrict__ mod,
                             float* __restrict__ h)
{
  __shared__ float red[256];
  int bt = blockIdx.x;
  int b  = bt / L_;
  int tid = threadIdx.x;
  const float* xr = x + (long)bt * D_;
  float s = 0.f;
  for (int d = tid; d < D_; d += 256) s += xr[d];
  red[tid] = s; __syncthreads();
  for (int off = 128; off > 0; off >>= 1) { if (tid < off) red[tid] += red[tid + off]; __syncthreads(); }
  float mean = red[0] / D_; __syncthreads();
  float v = 0.f;
  for (int d = tid; d < D_; d += 256) { float dd = xr[d] - mean; v += dd * dd; }
  red[tid] = v; __syncthreads();
  for (int off = 128; off > 0; off >>= 1) { if (tid < off) red[tid] += red[tid + off]; __syncthreads(); }
  float rstd = rsqrtf(red[0] / D_ + 1e-5f);
  const float* mr = mod + (long)b * 3 * D_;
  for (int d = tid; d < D_; d += 256) {
    float shift = mr[d], scale = mr[D_ + d];
    h[(long)bt * D_ + d] = (xr[d] - mean) * rstd * (1.f + scale) + shift;
  }
}

__global__ void softplus_kernel(float* __restrict__ delta, const float* __restrict__ bias, long n)
{
  long i = (long)blockIdx.x * 256 + threadIdx.x;
  if (i < n) {
    float v = delta[i] + bias[i % DI_];
    delta[i] = (v > 20.f) ? v : log1pf(__expf(v));
  }
}

// Selective-scan, one thread per (b, channel). Fuses +x*D and *silu(z).
__global__ void ssm_scan_kernel(const float* __restrict__ xz, const float* __restrict__ dbc,
                                const float* __restrict__ delta,
                                const float* __restrict__ A_log, const float* __restrict__ Dp,
                                float* __restrict__ yg, int reverse)
{
  __shared__ float sB[DS_], sC[DS_];
  int d = blockIdx.x * 256 + threadIdx.x;   // 0..DI-1
  int b = blockIdx.y;
  float A[DS_];
  #pragma unroll
  for (int s = 0; s < DS_; ++s) A[s] = -__expf(A_log[d * DS_ + s]);
  float Dv = Dp[d];
  float hst[DS_];
  #pragma unroll
  for (int s = 0; s < DS_; ++s) hst[s] = 0.f;

  for (int step = 0; step < L_; ++step) {
    int t = reverse ? (L_ - 1 - step) : step;
    long row = (long)(b * L_ + t);
    __syncthreads();
    if (threadIdx.x < 32) {
      float v = dbc[row * 64 + 32 + threadIdx.x];
      if (threadIdx.x < 16) sB[threadIdx.x] = v; else sC[threadIdx.x - 16] = v;
    }
    __syncthreads();
    float del = delta[row * DI_ + d];
    float xv  = xz[row * 2 * DI_ + d];
    float zv  = xz[row * 2 * DI_ + DI_ + d];
    float y = 0.f;
    #pragma unroll
    for (int s = 0; s < DS_; ++s) {
      float dA = __expf(del * A[s]);
      hst[s] = dA * hst[s] + del * sB[s] * xv;
      y += hst[s] * sC[s];
    }
    y += xv * Dv;
    y *= zv / (1.f + __expf(-zv));          // * silu(z)
    yg[row * DI_ + d] = y;
  }
}

__global__ void silu_mul_kernel(float* __restrict__ u, const float* __restrict__ v, long n)
{
  long i = (long)blockIdx.x * 256 + threadIdx.x;
  if (i < n) { float a = u[i]; u[i] = a / (1.f + __expf(-a)) * v[i]; }
}

// x += gate * (y1 [+ y2])
__global__ void resid_kernel(float* __restrict__ x, const float* __restrict__ mod,
                             const float* __restrict__ y1, const float* __restrict__ y2)
{
  int bt = blockIdx.x;
  int b  = bt / L_;
  const float* g = mod + (long)b * 3 * D_ + 2 * D_;
  for (int d = threadIdx.x; d < D_; d += 256) {
    float y = y1[(long)bt * D_ + d];
    if (y2) y += y2[(long)bt * D_ + d];
    x[(long)bt * D_ + d] += g[d] * y;
  }
}

__global__ void softmax_kernel(float* __restrict__ scores)
{
  __shared__ float red[256];
  long row = blockIdx.x;
  float* r = scores + row * (long)L_;
  int tid = threadIdx.x;
  float m = -1e30f;
  for (int i = tid; i < L_; i += 256) m = fmaxf(m, r[i]);
  red[tid] = m; __syncthreads();
  for (int off = 128; off > 0; off >>= 1) { if (tid < off) red[tid] = fmaxf(red[tid], red[tid + off]); __syncthreads(); }
  m = red[0]; __syncthreads();
  float s = 0.f;
  for (int i = tid; i < L_; i += 256) { float e = __expf(r[i] - m); r[i] = e; s += e; }
  red[tid] = s; __syncthreads();
  for (int off = 128; off > 0; off >>= 1) { if (tid < off) red[tid] += red[tid + off]; __syncthreads(); }
  float inv = 1.f / red[0];
  for (int i = tid; i < L_; i += 256) r[i] *= inv;
}

// =====================================================================
// Host side
// =====================================================================
static void gemm(hipStream_t st, const float* A, const float* B, float* C,
                 int M, int N, int K, int lda, int ldb, int ldc, int transB, float alpha,
                 int nb = 1, int inner = 1,
                 long offA = 0, long offB = 0, long offC = 0,
                 long sAo = 0, long sAi = 0, long sBo = 0, long sBi = 0,
                 long sCo = 0, long sCi = 0)
{
  dim3 grid(N / 64, M / 128, nb);
  if (transB)
    wmma_gemm_kernel<1><<<grid, 256, 0, st>>>(A, B, C, K, lda, ldb, ldc, alpha,
                                              inner, offA, offB, offC,
                                              sAo, sAi, sBo, sBi, sCo, sCi);
  else
    wmma_gemm_kernel<0><<<grid, 256, 0, st>>>(A, B, C, K, lda, ldb, ldc, alpha,
                                              inner, offA, offB, offC,
                                              sAo, sAi, sBo, sBi, sCo, sCi);
}

// Input index map (jax tree-flatten: alphabetical dict keys; input_ids first, t last)
// mamba block (21 leaves): mlp.w1, mlp.w2, mlp.w3, mod_m_b, mod_m_w, mod_p_b, mod_p_w,
//   ssm_b.{A_log,D,dt_bias,dt_proj,in_proj,out_proj,x_proj}, ssm_f.{same}
// attn block (9 leaves): mlp.w1, mlp.w2, mlp.w3, mod_a_b, mod_a_w, mod_p_b, mod_p_w, out_proj, qkv
extern "C" void kernel_launch(void* const* d_in, const int* in_sizes, int n_in,
                              void* d_out, int out_size, void* d_ws, size_t ws_size,
                              hipStream_t stream)
{
  (void)in_sizes; (void)n_in; (void)out_size; (void)ws_size;
  auto F = [&](int i) { return (const float*)d_in[i]; };
  const int* ids = (const int*)d_in[0];

  const int base0 = 1;                 // blocks[0] (mamba, 21 leaves)
  const int base1 = base0 + 21;        // blocks[1] (attn, 9)
  const int base2 = base1 + 9;         // blocks[2] (mamba, 21)
  const int base3 = base2 + 21;        // blocks[3] (attn, 9)
  const int gbase = base3 + 9;         // 61
  const float* out_mod_b = F(gbase + 0);
  const float* out_mod_w = F(gbase + 1);
  const float* pos_emb   = F(gbase + 2);
  const float* t_b1      = F(gbase + 3);
  const float* t_b2      = F(gbase + 4);
  const float* t_w1      = F(gbase + 5);
  const float* t_w2      = F(gbase + 6);
  const float* tok_emb   = F(gbase + 7);
  const float* t_in      = F(gbase + 8);

  // ---- workspace carve (fp32 elements) ----
  float* ws = (float*)d_ws;
  size_t p = 0;
  auto alloc = [&](size_t n) { float* r = ws + p; p += n; return r; };
  float* c    = alloc(B_ * COND_);
  float* mod  = alloc(B_ * 3 * D_);
  float* x    = alloc((size_t)B_ * L_ * D_);
  float* h    = alloc((size_t)B_ * L_ * D_);
  float* xzf  = alloc((size_t)B_ * L_ * 2 * DI_);
  float* xzb  = alloc((size_t)B_ * L_ * 2 * DI_);
  float* dbcf = alloc((size_t)B_ * L_ * 64);
  float* dbcb = alloc((size_t)B_ * L_ * 64);
  float* delf = alloc((size_t)B_ * L_ * DI_);
  float* delb = alloc((size_t)B_ * L_ * DI_);
  float* ygf  = alloc((size_t)B_ * L_ * DI_);
  float* ygb  = alloc((size_t)B_ * L_ * DI_);
  float* y1   = alloc((size_t)B_ * L_ * D_);
  float* y2   = alloc((size_t)B_ * L_ * D_);
  float* u    = alloc((size_t)B_ * L_ * HID_);
  float* vv   = alloc((size_t)B_ * L_ * HID_);
  float* qkv  = alloc((size_t)B_ * L_ * 3 * D_);
  float* sc   = alloc((size_t)B_ * NH_ * L_ * L_);

  const int M = B_ * L_;               // 1024 token rows

  // ---- embed + timestep conditioning ----
  embed_kernel<<<B_ * L_, 256, 0, stream>>>(ids, tok_emb, pos_emb, x);
  timestep_kernel<<<B_, COND_, 0, stream>>>(t_in, t_w1, t_b1, t_w2, t_b2, c);

  auto run_mlp = [&](int bb) {  // mod_p adaLN + SwiGLU + gated residual
    mod_kernel<<<dim3(3 * D_ / 256, B_), 256, 0, stream>>>(c, F(bb + 6), F(bb + 5), mod);
    adaln_kernel<<<B_ * L_, 256, 0, stream>>>(x, mod, h);
    gemm(stream, h, F(bb + 0), u,  M, HID_, D_, D_, HID_, HID_, 0, 1.f);   // w1
    gemm(stream, h, F(bb + 1), vv, M, HID_, D_, D_, HID_, HID_, 0, 1.f);   // w2
    long nh = (long)M * HID_;
    silu_mul_kernel<<<(nh + 255) / 256, 256, 0, stream>>>(u, vv, nh);
    gemm(stream, u, F(bb + 2), y1, M, D_, HID_, HID_, D_, D_, 0, 1.f);     // w3
    resid_kernel<<<B_ * L_, 256, 0, stream>>>(x, mod, y1, (const float*)nullptr);
  };

  auto run_ssm = [&](int sb, float* xz, float* dbc, float* del, float* yg,
                     float* yout, int rev) {
    // sb -> A_log, D, dt_bias, dt_proj, in_proj, out_proj, x_proj
    gemm(stream, h, F(sb + 4), xz, M, 2 * DI_, D_, D_, 2 * DI_, 2 * DI_, 0, 1.f); // in_proj
    gemm(stream, xz, F(sb + 6), dbc, M, 64, DI_, 2 * DI_, 64, 64, 0, 1.f);        // x_proj (x half)
    gemm(stream, dbc, F(sb + 3), del, M, DI_, DT_, 64, DI_, DI_, 0, 1.f);         // dt_proj
    long nd = (long)M * DI_;
    softplus_kernel<<<(nd + 255) / 256, 256, 0, stream>>>(del, F(sb + 2), nd);
    ssm_scan_kernel<<<dim3(DI_ / 256, B_), 256, 0, stream>>>(xz, dbc, del,
                                                             F(sb + 0), F(sb + 1), yg, rev);
    gemm(stream, yg, F(sb + 5), yout, M, D_, DI_, DI_, D_, D_, 0, 1.f);           // out_proj
  };

  auto mamba_block = [&](int bb) {
    mod_kernel<<<dim3(3 * D_ / 256, B_), 256, 0, stream>>>(c, F(bb + 4), F(bb + 3), mod);
    adaln_kernel<<<B_ * L_, 256, 0, stream>>>(x, mod, h);
    run_ssm(bb + 14, xzf, dbcf, delf, ygf, y1, 0);   // ssm_f (forward scan)
    run_ssm(bb + 7,  xzb, dbcb, delb, ygb, y2, 1);   // ssm_b (reverse scan == flip/ssm/flip)
    resid_kernel<<<B_ * L_, 256, 0, stream>>>(x, mod, y1, y2);
    run_mlp(bb);
  };

  auto attn_block = [&](int bb) {
    mod_kernel<<<dim3(3 * D_ / 256, B_), 256, 0, stream>>>(c, F(bb + 4), F(bb + 3), mod);
    adaln_kernel<<<B_ * L_, 256, 0, stream>>>(x, mod, h);
    gemm(stream, h, F(bb + 8), qkv, M, 3 * D_, D_, D_, 3 * D_, 3 * D_, 0, 1.f);   // qkv
    // scores[b,h] = q @ k^T / sqrt(HD)    (NT, batched over b*NH)
    gemm(stream, qkv, qkv, sc, L_, L_, HD_, 3 * D_, 3 * D_, L_, 1, 0.125f,
         B_ * NH_, NH_,
         /*offA=*/0, /*offB=*/D_, /*offC=*/0,
         (long)L_ * 3 * D_, HD_, (long)L_ * 3 * D_, HD_,
         (long)NH_ * L_ * L_, (long)L_ * L_);
    softmax_kernel<<<B_ * NH_ * L_, 256, 0, stream>>>(sc);
    // o[b,h] = attn @ v    (NN, batched), written into [B,L,NH,HD] layout
    gemm(stream, sc, qkv, y1, L_, HD_, L_, L_, 3 * D_, D_, 0, 1.f,
         B_ * NH_, NH_,
         0, 2 * D_, 0,
         (long)NH_ * L_ * L_, (long)L_ * L_, (long)L_ * 3 * D_, HD_,
         (long)L_ * D_, HD_);
    gemm(stream, y1, F(bb + 7), y2, M, D_, D_, D_, D_, D_, 0, 1.f);               // out_proj
    resid_kernel<<<B_ * L_, 256, 0, stream>>>(x, mod, y2, (const float*)nullptr);
    run_mlp(bb);
  };

  mamba_block(base0);
  attn_block(base1);
  mamba_block(base2);
  attn_block(base3);

  // ---- final adaLN + vocab projection (NT against tok_emb) ----
  mod_kernel<<<dim3(3 * D_ / 256, B_), 256, 0, stream>>>(c, out_mod_w, out_mod_b, mod);
  adaln_kernel<<<B_ * L_, 256, 0, stream>>>(x, mod, h);
  gemm(stream, h, tok_emb, (float*)d_out, M, V_, D_, D_, D_, V_, 1, 1.f);
}
